// DLRM_88716844466667
// MI455X (gfx1250) — matrix operand order repaired and checked by reading, not compiled, not run on hardware
//
#include <hip/hip_runtime.h>
#include <cstdint>

typedef uint32_t u32;
typedef __attribute__((ext_vector_type(16))) _Float16 v16h;
typedef __attribute__((ext_vector_type(8)))  float    v8f;

union AF { u32 u[8]; v16h h; };

// ---------------- WMMA fragment helpers (16x16x32 f16, f32 acc) ----------------
// A fragment from a row-major f16 tile (tileRow0 = first row of this wave's 16-row tile).
// Layout: lane<16 -> M=lane, K elems {0..15}; lane>=16 -> K {16..31}.
__device__ __forceinline__ v16h load_a16(const _Float16* tileRow0, int stride, int kt, int lane) {
    int m = lane & 15, kh = lane >> 4;
    const u32* q = (const u32*)(tileRow0 + m * stride + kt * 32 + kh * 16);
    AF f;
#pragma unroll
    for (int i = 0; i < 4; ++i) f.u[i] = q[i];       // halves 0..7   -> K = 0..7   (+16*kh)
#pragma unroll
    for (int i = 4; i < 8; ++i) f.u[i] = q[i];       // halves 8..15  -> K = 8..15  (+16*kh)
    return f.h;
}
// B fragment from N-major f16 weight (wT[n*stride + k]). lane%16 = N, elem i -> K = i + 16*kh.
__device__ __forceinline__ v16h load_b16(const _Float16* wT, int stride, int nt, int kt, int lane) {
    int n = (lane & 15) + nt * 16, kh = lane >> 4;
    const u32* q = (const u32*)(wT + n * stride + kt * 32 + kh * 16);
    AF f;
#pragma unroll
    for (int i = 0; i < 8; ++i) f.u[i] = q[i];
    return f.h;
}
__device__ __forceinline__ v8f wmma32(v8f acc, v16h a, v16h b) {
    return __builtin_amdgcn_wmma_f32_16x16x32_f16(false, a, false, b, (short)0, acc, false, false);
}
__device__ __forceinline__ float sigmoidf(float x) { return 1.f / (1.f + __expf(-x)); }

// ---------------- K0: weight convert f32 [din][dout] -> f16 [dout][KP] (K padded, zero fill) ----
__global__ void conv_w_kernel(const float* __restrict__ w, _Float16* __restrict__ wT,
                              int din, int dout, int KP) {
    int i = blockIdx.x * blockDim.x + threadIdx.x;
    if (i >= dout * KP) return;
    int n = i / KP, k = i % KP;
    wT[i] = (k < din) ? (_Float16)w[k * dout + n] : (_Float16)0.f;
}

// ---------------- K1: gathers + bottom MLP + genre proj (VALU, tiny FLOPs) ----------------
__global__ __launch_bounds__(64) void base_feats(
    const int* __restrict__ uid, const int* __restrict__ mid,
    const float* __restrict__ dense, const float* __restrict__ genres,
    const float* __restrict__ uemb, const float* __restrict__ iemb,
    const float* __restrict__ bw0, const float* __restrict__ bb0,
    const float* __restrict__ bw1, const float* __restrict__ bb1,
    const float* __restrict__ gw,  const float* __restrict__ gb,
    float* __restrict__ x0, float* __restrict__ usin, float* __restrict__ itin) {
    __shared__ float h1[64][129];
    int t = threadIdx.x;
    int s = blockIdx.x * 64 + t;
    float d[11];
#pragma unroll
    for (int j = 0; j < 11; ++j) d[j] = dense[s * 11 + j];
    for (int k = 0; k < 128; ++k) {
        float a = bb0[k];
#pragma unroll
        for (int j = 0; j < 11; ++j) a += d[j] * bw0[j * 128 + k];
        h1[t][k] = fmaxf(a, 0.f);
    }
    int ub = uid[s] * 24, ib = mid[s] * 24;
    for (int dd = 0; dd < 24; ++dd) {
        float ue = uemb[ub + dd], ie = iemb[ib + dd];
        x0[s * 144 + dd] = ue;       x0[s * 144 + 24 + dd] = ie;
        usin[s * 72 + dd] = ue;      itin[s * 72 + dd] = ie;
        float a = bb1[dd];
        for (int k = 0; k < 128; ++k) a += h1[t][k] * bw1[k * 24 + dd];
        a = fmaxf(a, 0.f);
        x0[s * 144 + 96 + dd] = a;   usin[s * 72 + 48 + dd] = a;
        float g = gb[dd];
#pragma unroll
        for (int gg = 0; gg < 19; ++gg) g += genres[s * 19 + gg] * gw[gg * 24 + dd];
        g = fmaxf(g, 0.f);
        x0[s * 144 + 120 + dd] = g;  itin[s * 72 + 48 + dd] = g;
    }
}

// ---------------- K2/K3: DIN attention pool (WMMA). One sample per workgroup round. ------------
__global__ __launch_bounds__(128) void din_pool(
    const int* __restrict__ histIdx, const float* __restrict__ ratings, int H, int padIdx,
    const float* __restrict__ embTab, const int* __restrict__ tgtIds, const float* __restrict__ tgtTab,
    const float* __restrict__ rpw, const float* __restrict__ rpb,
    const _Float16* __restrict__ w1T, const float* __restrict__ b1,
    const float* __restrict__ w2, const float* __restrict__ b2,
    float* __restrict__ x0, int x0off, float* __restrict__ sin_, int spb) {
    __shared__ _Float16 A[64 * 160];
    __shared__ float HE[64 * 24];
    __shared__ float tgt[48];
    __shared__ float logitS[64];
    __shared__ int   padF[64];
    __shared__ float red[2];
    int tid = threadIdx.x, lane = tid & 31, wv = tid >> 5;
    int MT = (H + 15) >> 4;
    for (int r = 0; r < spb; ++r) {
        int s = blockIdx.x * spb + r;
        if (tid < 24) { float tv = tgtTab[tgtIds[s] * 24 + tid]; tgt[tid] = tv; tgt[24 + tid] = tv; }
        __syncthreads();
        if (tid < H) {
            int j = tid;
            int idx = histIdx[s * H + j];
            float rr = ratings[s * H + j];
            padF[j] = (idx == padIdx);
            _Float16* row = A + j * 160;
#pragma unroll
            for (int d = 0; d < 24; ++d) {
                float he = embTab[idx * 24 + d];
                float rp = rpw[d] * rr + rpb[d];
                float t0 = tgt[d];
                HE[j * 24 + d] = he;
                row[d]       = (_Float16)he;        row[24 + d]  = (_Float16)rp;
                row[48 + d]  = (_Float16)t0;        row[72 + d]  = (_Float16)t0;
                row[96 + d]  = (_Float16)(he * t0); row[120 + d] = (_Float16)(rp * t0);
            }
#pragma unroll
            for (int d = 144; d < 160; ++d) row[d] = (_Float16)0.f;
        }
        __syncthreads();
        if (wv < MT) {  // wave wv handles rows [wv*16, wv*16+16)
            float part[8] = {0, 0, 0, 0, 0, 0, 0, 0};
            int n0 = lane & 15, kh = lane >> 4;
            for (int nt = 0; nt < 4; ++nt) {
                v8f acc = {0.f, 0.f, 0.f, 0.f, 0.f, 0.f, 0.f, 0.f};
#pragma unroll
                for (int kt = 0; kt < 5; ++kt)
                    acc = wmma32(acc, load_a16(A + wv * 16 * 160, 160, kt, lane),
                                      load_b16(w1T, 160, nt, kt, lane));
                int n = n0 + nt * 16;
                float bb = b1[n], ww = w2[n];
#pragma unroll
                for (int r2 = 0; r2 < 8; ++r2) part[r2] += fmaxf(acc[r2] + bb, 0.f) * ww;
            }
#pragma unroll
            for (int mk = 1; mk < 16; mk <<= 1)
#pragma unroll
                for (int r2 = 0; r2 < 8; ++r2) part[r2] += __shfl_xor(part[r2], mk, 32);
            if (n0 == 0) {
                float bb2 = b2[0];
#pragma unroll
                for (int r2 = 0; r2 < 8; ++r2) logitS[wv * 16 + r2 + 8 * kh] = part[r2] + bb2;
            }
        }
        __syncthreads();
        if (tid == 0) {
            float mx = -1e30f;
            for (int j = 0; j < H; ++j) {
                float l = padF[j] ? -10000.f : logitS[j];
                logitS[j] = l; mx = fmaxf(mx, l);
            }
            float sum = 0.f;
            for (int j = 0; j < H; ++j) sum += __expf(logitS[j] - mx);
            red[0] = mx; red[1] = 1.f / sum;
        }
        __syncthreads();
        if (tid < 24) {
            float acc = 0.f, mx = red[0], inv = red[1];
            for (int j = 0; j < H; ++j) acc += __expf(logitS[j] - mx) * inv * HE[j * 24 + tid];
            x0[s * 144 + x0off + tid] = acc;
            sin_[s * 72 + 24 + tid] = acc;
        }
        __syncthreads();
    }
}

// ---------------- K4: cross stack + streams + top MLP (WMMA), 32 rows / WG -------------------
struct MainParams {
    const float *x0, *usin, *itin;
    const _Float16 *cwT, *cgT;                 // 3 stacked [144*160] each
    const float *cb0, *cb1, *cb2, *gb0, *gb1, *gb2;
    const _Float16 *us0T, *us1T, *it0T, *it1T, *t0T, *t1T, *t2T;
    const float *us0b, *us1b, *it0b, *it1b, *t0b, *t1b, *t2b, *t3w, *t3b;
    float* out;
};

__device__ __forceinline__ void layer16(const _Float16* inRow, int KP,
                                        const _Float16* wT, const float* b, int N,
                                        _Float16* outBase, int strideOut,
                                        int wv, int lane) {
    int NT = N >> 4, KT = KP >> 5;
    int n0 = lane & 15, kh = lane >> 4;
    for (int nt = 0; nt < NT; ++nt) {
        v8f acc = {0.f, 0.f, 0.f, 0.f, 0.f, 0.f, 0.f, 0.f};
        for (int kt = 0; kt < KT; ++kt)
            acc = wmma32(acc, load_a16(inRow, KP, kt, lane), load_b16(wT, KP, nt, kt, lane));
        int n = n0 + nt * 16;
        float bb = b[n];
#pragma unroll
        for (int r2 = 0; r2 < 8; ++r2) {
            int m = wv * 16 + r2 + 8 * kh;
            outBase[m * strideOut + n] = (_Float16)fmaxf(acc[r2] + bb, 0.f);
        }
    }
}

__global__ __launch_bounds__(64) void fused_main(MainParams p) {
    __shared__ __align__(16) char smem[65536];
    _Float16* smA  = (_Float16*)(smem);           // [32][160] f16  (cross ping; later t2 [32][128])
    _Float16* smB  = (_Float16*)(smem + 10240);   // [32][160] f16  (cross pong; holds x3)
    float*    x0f  = (float*)   (smem + 20480);   // [32][144] f32  (cross phase only)
    _Float16* htop = (_Float16*)(smem + 20480);   // [32][352] f16  (after cross)
    _Float16* smT1 = (_Float16*)(smem + 43008);   // [32][256] f16  (h1 [32][128] / t1 [32][256])
    _Float16* smSI = (_Float16*)(smem + 59392);   // [32][96]  f16  (stream input)
    int tid = threadIdx.x, lane = tid & 31, wv = tid >> 5;  // 2 waves
    int rowbase = blockIdx.x * 32;

    // P0: stage x0 tile (f32 + f16, zero K-pad)
    for (int i = tid; i < 32 * 144; i += 64) {
        int rr = i / 144, c = i % 144;
        float v = p.x0[(rowbase + rr) * 144 + c];
        x0f[i] = v;
        smA[rr * 160 + c] = (_Float16)v;
    }
    for (int i = tid; i < 32 * 16; i += 64) {
        int rr = i / 16, c = 144 + (i % 16);
        smA[rr * 160 + c] = (_Float16)0.f; smB[rr * 160 + c] = (_Float16)0.f;
    }
    __syncthreads();

    // Cross stack: 3 gated steps, wave-local rows.
    const float* cbs[3] = {p.cb0, p.cb1, p.cb2};
    const float* gbs[3] = {p.gb0, p.gb1, p.gb2};
    _Float16 *src = smA, *dst = smB;
    int n0 = lane & 15, kh = lane >> 4;
    for (int step = 0; step < 3; ++step) {
        const _Float16* W = p.cwT + step * 144 * 160;
        const _Float16* G = p.cgT + step * 144 * 160;
        const float* vb = cbs[step];
        const float* gb = gbs[step];
        const _Float16* srow = src + wv * 16 * 160;
        for (int nt = 0; nt < 9; ++nt) {
            v8f av = {0.f, 0.f, 0.f, 0.f, 0.f, 0.f, 0.f, 0.f};
            v8f ag = {0.f, 0.f, 0.f, 0.f, 0.f, 0.f, 0.f, 0.f};
#pragma unroll
            for (int kt = 0; kt < 5; ++kt) {
                v16h a = load_a16(srow, 160, kt, lane);
                av = wmma32(av, a, load_b16(W, 160, nt, kt, lane));
                ag = wmma32(ag, a, load_b16(G, 160, nt, kt, lane));
            }
            int n = n0 + nt * 16;
            float vbn = vb[n], gbn = gb[n];
#pragma unroll
            for (int r2 = 0; r2 < 8; ++r2) {
                int m = wv * 16 + r2 + 8 * kh;
                float v = av[r2] + vbn;
                float g = sigmoidf(ag[r2] + gbn);
                float c = x0f[m * 144 + n] * v;
                float xp = (float)src[m * 160 + n];
                dst[m * 160 + n] = (_Float16)(g * c + (1.f - g) * xp);
            }
        }
        _Float16* t = src; src = dst; dst = t;
    }
    // x3 now in smB (== src).
    __syncthreads();   // x0f region becomes htop

    // User stream: 72->128->64 (relu), keep us in regs for bilinear.
    for (int i = lane; i < 16 * 96; i += 32) {
        int rr = i / 96, c = i % 96, m = wv * 16 + rr;
        smSI[m * 96 + c] = (c < 72) ? (_Float16)p.usin[(rowbase + m) * 72 + c] : (_Float16)0.f;
    }
    layer16(smSI + wv * 16 * 96, 96, p.us0T, p.us0b, 128, smT1, 128, wv, lane);
    float usv[4][8];
    for (int nt = 0; nt < 4; ++nt) {
        v8f acc = {0.f, 0.f, 0.f, 0.f, 0.f, 0.f, 0.f, 0.f};
#pragma unroll
        for (int kt = 0; kt < 4; ++kt)
            acc = wmma32(acc, load_a16(smT1 + wv * 16 * 128, 128, kt, lane),
                              load_b16(p.us1T, 128, nt, kt, lane));
        float bb = p.us1b[n0 + nt * 16];
#pragma unroll
        for (int r2 = 0; r2 < 8; ++r2) {
            float v = fmaxf(acc[r2] + bb, 0.f);
            usv[nt][r2] = v;
            int m = wv * 16 + r2 + 8 * kh;
            htop[m * 352 + 144 + n0 + nt * 16] = (_Float16)v;
        }
    }
    // Item stream + bilinear.
    for (int i = lane; i < 16 * 96; i += 32) {
        int rr = i / 96, c = i % 96, m = wv * 16 + rr;
        smSI[m * 96 + c] = (c < 72) ? (_Float16)p.itin[(rowbase + m) * 72 + c] : (_Float16)0.f;
    }
    layer16(smSI + wv * 16 * 96, 96, p.it0T, p.it0b, 128, smT1, 128, wv, lane);
    for (int nt = 0; nt < 4; ++nt) {
        v8f acc = {0.f, 0.f, 0.f, 0.f, 0.f, 0.f, 0.f, 0.f};
#pragma unroll
        for (int kt = 0; kt < 4; ++kt)
            acc = wmma32(acc, load_a16(smT1 + wv * 16 * 128, 128, kt, lane),
                              load_b16(p.it1T, 128, nt, kt, lane));
        float bb = p.it1b[n0 + nt * 16];
#pragma unroll
        for (int r2 = 0; r2 < 8; ++r2) {
            float v = fmaxf(acc[r2] + bb, 0.f);
            int m = wv * 16 + r2 + 8 * kh;
            htop[m * 352 + 208 + n0 + nt * 16] = (_Float16)v;
            htop[m * 352 + 272 + n0 + nt * 16] = (_Float16)(v * usv[nt][r2]);
        }
    }
    // Copy x3 into htop[:,0:144], zero K-pad cols.
    for (int i = lane; i < 16 * 144; i += 32) {
        int rr = i / 144, c = i % 144, m = wv * 16 + rr;
        htop[m * 352 + c] = src[m * 160 + c];
    }
    for (int i = lane; i < 16 * 16; i += 32) {
        int rr = i / 16, c = 336 + (i % 16);
        htop[(wv * 16 + rr) * 352 + c] = (_Float16)0.f;
    }
    __syncthreads();   // smT1 stride changes 128->256 across waves

    // Top MLP: 352->256->128->64->1
    layer16(htop + wv * 16 * 352, 352, p.t0T, p.t0b, 256, smT1, 256, wv, lane);
    _Float16* t2 = smA;  // reuse, dead after cross
    layer16(smT1 + wv * 16 * 256, 256, p.t1T, p.t1b, 128, t2, 128, wv, lane);
    {
        float part[8] = {0, 0, 0, 0, 0, 0, 0, 0};
        for (int nt = 0; nt < 4; ++nt) {
            v8f acc = {0.f, 0.f, 0.f, 0.f, 0.f, 0.f, 0.f, 0.f};
#pragma unroll
            for (int kt = 0; kt < 4; ++kt)
                acc = wmma32(acc, load_a16(t2 + wv * 16 * 128, 128, kt, lane),
                                  load_b16(p.t2T, 128, nt, kt, lane));
            int n = n0 + nt * 16;
            float bb = p.t2b[n], ww = p.t3w[n];
#pragma unroll
            for (int r2 = 0; r2 < 8; ++r2) part[r2] += fmaxf(acc[r2] + bb, 0.f) * ww;
        }
#pragma unroll
        for (int mk = 1; mk < 16; mk <<= 1)
#pragma unroll
            for (int r2 = 0; r2 < 8; ++r2) part[r2] += __shfl_xor(part[r2], mk, 32);
        if (n0 == 0) {
            float b3 = p.t3b[0];
#pragma unroll
            for (int r2 = 0; r2 < 8; ++r2)
                p.out[rowbase + wv * 16 + r2 + 8 * kh] = part[r2] + b3;
        }
    }
}

// ---------------- host launch ----------------
// d_in flatten assumption: 8 input arrays in dict order, then params tree-flattened
// with dict keys sorted (JAX pytree order), lists in order, {b,w} per linear.
enum {
    IN_UID = 0, IN_MID, IN_DENSE, IN_HIST, IN_HISTR, IN_GENRES, IN_IHIST, IN_IHISTR,
    P_BM0B = 8, P_BM0W, P_BM1B, P_BM1W,
    P_CB1 = 12, P_CB2, P_CB3,
    P_CG1B = 15, P_CG1W, P_CG2B, P_CG2W, P_CG3B, P_CG3W,
    P_CW1 = 21, P_CW2, P_CW3,
    P_DA0B = 24, P_DA0W, P_DA1B, P_DA1W,
    P_GPB = 28, P_GPW,
    P_HISTEMB = 30,
    P_IDA0B = 31, P_IDA0W, P_IDA1B, P_IDA1W,
    P_ITEMEMB = 35,
    P_IS0B = 36, P_IS0W, P_IS1B, P_IS1W,
    P_RATEREMB = 40,
    P_RPB = 41, P_RPW,
    P_TM0B = 43, P_TM0W, P_TM1B, P_TM1W, P_TM2B, P_TM2W, P_TM3B, P_TM3W,
    P_USEREMB = 51,
    P_US0B = 52, P_US0W, P_US1B, P_US1W
};

#define BATCH 16384

extern "C" void kernel_launch(void* const* d_in, const int* in_sizes, int n_in,
                              void* d_out, int out_size, void* d_ws, size_t ws_size,
                              hipStream_t stream) {
    (void)in_sizes; (void)n_in; (void)out_size; (void)ws_size;
    char* ws = (char*)d_ws;
    // f16 weight region offsets (bytes)
    const size_t O_DINW  = 0;                            // 64*160*2   = 20480
    const size_t O_IDINW = O_DINW + 20480;               // 20480
    const size_t O_CWT   = O_IDINW + 20480;              // 3*144*160*2 = 138240
    const size_t O_CGT   = O_CWT + 138240;               // 138240
    const size_t O_US0   = O_CGT + 138240;               // 128*96*2 = 24576
    const size_t O_US1   = O_US0 + 24576;                // 64*128*2 = 16384
    const size_t O_IT0   = O_US1 + 16384;                // 24576
    const size_t O_IT1   = O_IT0 + 24576;                // 16384
    const size_t O_T0    = O_IT1 + 16384;                // 256*352*2 = 180224
    const size_t O_T1    = O_T0 + 180224;                // 128*256*2 = 65536
    const size_t O_T2    = O_T1 + 65536;                 // 64*128*2 = 16384
    const size_t O_X0    = O_T2 + 16384;                 // B*144*4
    const size_t O_USIN  = O_X0 + (size_t)BATCH * 144 * 4;
    const size_t O_ITIN  = O_USIN + (size_t)BATCH * 72 * 4;

    auto F  = [&](int i) { return (const float*)d_in[i]; };
    auto I  = [&](int i) { return (const int*)d_in[i]; };
    auto H16 = [&](size_t off) { return (_Float16*)(ws + off); };

    auto conv = [&](int widx, size_t off, int din, int dout, int KP) {
        int tot = dout * KP;
        conv_w_kernel<<<(tot + 255) / 256, 256, 0, stream>>>(F(widx), H16(off), din, dout, KP);
    };
    // K0: convert all GEMM weights
    conv(P_DA0W,  O_DINW,  144, 64, 160);
    conv(P_IDA0W, O_IDINW, 144, 64, 160);
    conv(P_CW1, O_CWT + 0 * 46080, 144, 144, 160);
    conv(P_CW2, O_CWT + 1 * 46080, 144, 144, 160);
    conv(P_CW3, O_CWT + 2 * 46080, 144, 144, 160);
    conv(P_CG1W, O_CGT + 0 * 46080, 144, 144, 160);
    conv(P_CG2W, O_CGT + 1 * 46080, 144, 144, 160);
    conv(P_CG3W, O_CGT + 2 * 46080, 144, 144, 160);
    conv(P_US0W, O_US0, 72, 128, 96);
    conv(P_US1W, O_US1, 128, 64, 128);
    conv(P_IS0W, O_IT0, 72, 128, 96);
    conv(P_IS1W, O_IT1, 128, 64, 128);
    conv(P_TM0W, O_T0, 336, 256, 352);
    conv(P_TM1W, O_T1, 256, 128, 256);
    conv(P_TM2W, O_T2, 128, 64, 128);

    float* x0   = (float*)(ws + O_X0);
    float* usin = (float*)(ws + O_USIN);
    float* itin = (float*)(ws + O_ITIN);

    // K1: base features
    base_feats<<<BATCH / 64, 64, 0, stream>>>(
        I(IN_UID), I(IN_MID), F(IN_DENSE), F(IN_GENRES),
        F(P_USEREMB), F(P_ITEMEMB),
        F(P_BM0W), F(P_BM0B), F(P_BM1W), F(P_BM1B),
        F(P_GPW), F(P_GPB), x0, usin, itin);

    // K2: user DIN pool (H=50, target = item embed)
    din_pool<<<2048, 128, 0, stream>>>(
        I(IN_HIST), F(IN_HISTR), 50, 62423,
        F(P_HISTEMB), I(IN_MID), F(P_ITEMEMB),
        F(P_RPW), F(P_RPB),
        H16(O_DINW), F(P_DA0B), F(P_DA1W), F(P_DA1B),
        x0, 48, usin, BATCH / 2048);

    // K3: item DIN pool (H=30, target = user embed)
    din_pool<<<2048, 128, 0, stream>>>(
        I(IN_IHIST), F(IN_IHISTR), 30, 162541,
        F(P_RATEREMB), I(IN_UID), F(P_USEREMB),
        F(P_RPW), F(P_RPB),
        H16(O_IDINW), F(P_IDA0B), F(P_IDA1W), F(P_IDA1B),
        x0, 72, itin, BATCH / 2048);

    // K4: cross + streams + top
    MainParams p;
    p.x0 = x0; p.usin = usin; p.itin = itin;
    p.cwT = H16(O_CWT); p.cgT = H16(O_CGT);
    p.cb0 = F(P_CB1); p.cb1 = F(P_CB2); p.cb2 = F(P_CB3);
    p.gb0 = F(P_CG1B); p.gb1 = F(P_CG2B); p.gb2 = F(P_CG3B);
    p.us0T = H16(O_US0); p.us1T = H16(O_US1);
    p.it0T = H16(O_IT0); p.it1T = H16(O_IT1);
    p.t0T = H16(O_T0); p.t1T = H16(O_T1); p.t2T = H16(O_T2);
    p.us0b = F(P_US0B); p.us1b = F(P_US1B);
    p.it0b = F(P_IS0B); p.it1b = F(P_IS1B);
    p.t0b = F(P_TM0B); p.t1b = F(P_TM1B); p.t2b = F(P_TM2B);
    p.t3w = F(P_TM3W); p.t3b = F(P_TM3B);
    p.out = (float*)d_out;
    fused_main<<<BATCH / 32, 64, 0, stream>>>(p);
}
